// EmotionModel_83116207112676
// MI455X (gfx1250) — matrix operand order repaired and checked by reading, not compiled, not run on hardware
//
#include <hip/hip_runtime.h>
#include <hip/hip_bf16.h>
#include <cstdint>

// Fused GRU-ish cell for MI455X (gfx1250, wave32).
// Phase 1: one-time fp32 -> packed bf16 conversion of input + 6 weight matrices into d_ws.
// Phase 2: fused 6-gate GEMM (input @ W[g]^T) with v_wmma_f32_16x16x32_bf16, fp32 accum,
//          double-buffered LDS fed by the Tensor Data Mover (tensor_load_to_lds),
//          then sigmoid gate math + output write.

typedef __bf16 v16bf __attribute__((ext_vector_type(16)));
typedef __bf16 v2bf  __attribute__((ext_vector_type(2)));
typedef float  v8f   __attribute__((ext_vector_type(8)));
typedef unsigned int u32x4 __attribute__((ext_vector_type(4)));
typedef int  i32x8 __attribute__((ext_vector_type(8)));
typedef int  i32x4 __attribute__((ext_vector_type(4)));

#define BATCH 8192
#define DK    4096
#define HN    4096
#define BM    128          // block tile rows: 4 waves x 2 subtiles of 16
#define BN    32           // block tile cols: 2 waves x 16
#define BK    32           // K step (bf16 WMMA K)
#define LDSW  20           // LDS row stride in dwords (16 data + 4 pad)
#define KT    (DK / BK)    // 128

#if __has_builtin(__builtin_amdgcn_tensor_load_to_lds) && __has_builtin(__builtin_amdgcn_s_wait_tensorcnt)
#define USE_TDM 1
#else
#define USE_TDM 0
#endif

__device__ __forceinline__ uint32_t pack_bf16(float lo, float hi) {
#if __has_builtin(__builtin_amdgcn_cvt_pk_bf16_f32)
    union { v2bf v; uint32_t u; } cv;
    cv.v = __builtin_amdgcn_cvt_pk_bf16_f32(lo, hi);
    return cv.u;
#else
    union { float f; uint32_t u; } a, b;
    a.f = lo; b.f = hi;
    uint32_t ua = a.u + 0x7FFFu + ((a.u >> 16) & 1u);
    uint32_t ub = b.u + 0x7FFFu + ((b.u >> 16) & 1u);
    return (ua >> 16) | (ub & 0xFFFF0000u);
#endif
}

__device__ __forceinline__ float sigmoidf_fast(float x) {
    return 1.0f / (1.0f + __expf(-x));
}

#if USE_TDM
// One 2-D TDM descriptor: tile of (tile_d1 rows) x (tile_d0 dwords) from a row-major
// tensor of width tensor_d0 dwords (stride0 dwords), DMA'd to LDS at lds_off with
// 4-dword padding appended after every 16 data dwords (matches LDSW=20 rows).
// This toolchain's builtin takes 6 args: (g0 u32x4, g1 i32x8, g2 i32x4, g3 i32x4, i32x8, cpol).
__device__ __forceinline__ void tdm_load_2d(uint32_t lds_off, const uint32_t* gptr,
                                            uint32_t tensor_d0, uint32_t tensor_d1,
                                            uint32_t tile_d0, uint32_t tile_d1,
                                            uint32_t stride0) {
    const uint64_t ga = (uint64_t)(uintptr_t)gptr;
    u32x4 g0;
    g0[0] = 1u;                                   // count=1 (valid user descriptor)
    g0[1] = lds_off;                              // LDS byte address
    g0[2] = (uint32_t)ga;                         // global_addr[31:0]
    g0[3] = ((uint32_t)(ga >> 32) & 0x01FFFFFFu)  // global_addr[56:32]
            | (2u << 30);                         // type = 2 ("image")
    i32x8 g1;
    g1[0] = (2 << 16)      // data_size = 4 bytes
          | (1 << 20)      // pad_enable
          | (3 << 22)      // pad_interval: 16 dwords
          | (3 << 25);     // pad_amount: 4 dwords
    g1[1] = (int)((tensor_d0 & 0xFFFFu) << 16);                       // tensor_dim0 lo
    g1[2] = (int)((tensor_d0 >> 16) | ((tensor_d1 & 0xFFFFu) << 16)); // dim0 hi | dim1 lo
    g1[3] = (int)((tensor_d1 >> 16) | (tile_d0 << 16));               // dim1 hi | tile_dim0
    g1[4] = (int)tile_d1;                                             // tile_dim1 (tile_dim2=0)
    g1[5] = (int)stride0;                                             // tensor_dim0_stride[31:0]
    g1[6] = 0;
    g1[7] = 0;
    const i32x4 z4 = {0, 0, 0, 0};
    const i32x8 z8 = {0, 0, 0, 0, 0, 0, 0, 0};
    __builtin_amdgcn_tensor_load_to_lds(g0, g1, z4, z4, z8, 0);
}
#endif

// ---------------- Phase 1: elementwise fp32 -> bf16 pack ----------------
__global__ __launch_bounds__(256)
void cvt_f32_bf16_kernel(const float4* __restrict__ src, uint2* __restrict__ dst, int n4) {
    int i = blockIdx.x * blockDim.x + threadIdx.x;
    const int stride = gridDim.x * blockDim.x;
    for (; i < n4; i += stride) {
        const float4 f = src[i];
        dst[i] = make_uint2(pack_bf16(f.x, f.y), pack_bf16(f.z, f.w));
    }
}

// ---------------- Phase 2: fused 6-gate WMMA GEMM ----------------
template<bool PRE>   // PRE: operands pre-converted to packed bf16 dwords
__global__ __launch_bounds__(256)
void gru_wmma_gemm(const void* __restrict__ inp_,
                   const void* __restrict__ w0_, const void* __restrict__ w1_,
                   const void* __restrict__ w2_, const void* __restrict__ w3_,
                   const void* __restrict__ w4_, const void* __restrict__ w5_,
                   const float* __restrict__ hs,
                   const float* __restrict__ b_ir, const float* __restrict__ b_iz,
                   const float* __restrict__ b_in, const float* __restrict__ b_hr,
                   const float* __restrict__ b_hz, const float* __restrict__ b_hn,
                   float* __restrict__ out)
{
    __shared__ __attribute__((aligned(16))) uint32_t Alds[2][BM][LDSW];
    __shared__ __attribute__((aligned(16))) uint32_t Wlds[2][6][BN][LDSW];

    const int t     = threadIdx.x;
    const int lane  = t & 31;
    const int wid   = t >> 5;          // 0..7
    const int waveM = wid & 3;         // 0..3 -> 32-row strip (two 16-row subtiles)
    const int waveN = wid >> 2;        // 0..1 -> 16-col subtile
    const int half  = lane >> 4;
    const int r     = lane & 15;

    const int bm0 = blockIdx.x * BM;
    const int bn0 = blockIdx.y * BN;

    const void* wv[6] = { w0_, w1_, w2_, w3_, w4_, w5_ };

    // ---- staging of one K tile into LDS buffer `buf` ----
    auto stage = [&](int kt, int buf) {
        const int k0 = kt * BK;
        if constexpr (PRE) {
#if USE_TDM
            // Single wave drives the Tensor Data Mover: 7 tile descriptors.
            if (wid == 0) {
                const uint32_t* A = (const uint32_t*)inp_;     // [BATCH][DK/2] dwords
                tdm_load_2d((uint32_t)(uintptr_t)&Alds[buf][0][0],
                            A + (size_t)bm0 * (DK / 2) + k0 / 2,
                            DK / 2, BATCH, BK / 2, BM, DK / 2);
#pragma unroll
                for (int g = 0; g < 6; ++g) {
                    const uint32_t* W = (const uint32_t*)wv[g];
                    tdm_load_2d((uint32_t)(uintptr_t)&Wlds[buf][g][0][0],
                                W + (size_t)bn0 * (DK / 2) + k0 / 2,
                                DK / 2, HN, BK / 2, BN, DK / 2);
                }
            }
#else
            const int ar = t >> 2, aq = t & 3;     // A uint4 rows ar, ar+64
            const int wr = t >> 3, wq = t & 7;     // W rows 0..31, uint2 slots
            const uint32_t* A = (const uint32_t*)inp_;
            const uint4 a0 = ((const uint4*)(A + (size_t)(bm0 + ar)      * (DK/2) + k0/2))[aq];
            const uint4 a1 = ((const uint4*)(A + (size_t)(bm0 + ar + 64) * (DK/2) + k0/2))[aq];
            *(uint4*)&Alds[buf][ar][aq * 4]      = a0;
            *(uint4*)&Alds[buf][ar + 64][aq * 4] = a1;
#pragma unroll
            for (int g = 0; g < 6; ++g) {
                const uint32_t* W = (const uint32_t*)wv[g];
                const uint2 w = ((const uint2*)(W + (size_t)(bn0 + wr) * (DK/2) + k0/2))[wq];
                *(uint2*)&Wlds[buf][g][wr][wq * 2] = w;
            }
            if (kt + 1 < KT) {
                __builtin_prefetch(A + (size_t)(bm0 + ar) * (DK/2) + k0/2 + BK/2, 0, 1);
                __builtin_prefetch((const uint32_t*)wv[0] + (size_t)(bn0 + wr) * (DK/2) + k0/2 + BK/2, 0, 1);
            }
#endif
        } else {
            // fallback (tiny d_ws): convert fp32 -> bf16 while staging
            const float* A = (const float*)inp_;
            const int frow = t >> 3;
            const int fq   = t & 7;
#pragma unroll
            for (int p = 0; p < 4; ++p) {
                const int row = frow + p * 32;
                const float4 f = *(const float4*)(A + (size_t)(bm0 + row) * DK + k0 + fq * 4);
                Alds[buf][row][fq * 2]     = pack_bf16(f.x, f.y);
                Alds[buf][row][fq * 2 + 1] = pack_bf16(f.z, f.w);
            }
#pragma unroll
            for (int g = 0; g < 6; ++g) {
                const float* W = (const float*)wv[g];
                const float4 f = *(const float4*)(W + (size_t)(bn0 + frow) * DK + k0 + fq * 4);
                Wlds[buf][g][frow][fq * 2]     = pack_bf16(f.x, f.y);
                Wlds[buf][g][frow][fq * 2 + 1] = pack_bf16(f.z, f.w);
            }
        }
    };

    v8f acc[6][2];
#pragma unroll
    for (int g = 0; g < 6; ++g)
#pragma unroll
        for (int s = 0; s < 2; ++s)
            acc[g][s] = (v8f){0.f,0.f,0.f,0.f,0.f,0.f,0.f,0.f};

    // prologue: stage tile 0 into buffer 0
    stage(0, 0);
    if constexpr (PRE) {
#if USE_TDM
        if (wid == 0) __builtin_amdgcn_s_wait_tensorcnt(0);
#endif
    }
    __syncthreads();

    union Frag { uint32_t u[8]; v16bf v; };

    for (int kt = 0; kt < KT; ++kt) {
        const int cur = kt & 1;
        const int knext = (kt + 1 < KT) ? kt + 1 : KT - 1;  // clamp; last re-stage unread

#if USE_TDM
        if constexpr (PRE) stage(knext, cur ^ 1);   // async DMA overlaps the WMMAs below
#endif

        // A fragments (ISA 16-bit A 16x32 layout): two 16-row subtiles
        Frag fa[2];
        const uint32_t* Ar0 = &Alds[cur][waveM * 32 + r][0];
        const uint32_t* Ar1 = &Alds[cur][waveM * 32 + 16 + r][0];
#pragma unroll
        for (int v = 0; v < 4; ++v) {
            fa[0].u[v]     = Ar0[4 * half + v];
            fa[0].u[v + 4] = Ar0[4 * half + 8 + v];
            fa[1].u[v]     = Ar1[4 * half + v];
            fa[1].u[v + 4] = Ar1[4 * half + 8 + v];
        }
        // all six B fragments up front (distinct regs -> loads pipelined ahead of WMMAs)
        Frag fb[6];
#pragma unroll
        for (int g = 0; g < 6; ++g) {
            const uint32_t* Wr = &Wlds[cur][g][waveN * 16 + r][0];
#pragma unroll
            for (int v = 0; v < 8; ++v) fb[g].u[v] = Wr[4 * half + v + 4 * (v >> 2)];
        }
        // 12 back-to-back WMMAs
#pragma unroll
        for (int g = 0; g < 6; ++g) {
            acc[g][0] = __builtin_amdgcn_wmma_f32_16x16x32_bf16(
                false, fa[0].v, false, fb[g].v, (short)0, acc[g][0], false, false);
            acc[g][1] = __builtin_amdgcn_wmma_f32_16x16x32_bf16(
                false, fa[1].v, false, fb[g].v, (short)0, acc[g][1], false, false);
        }

        if constexpr (PRE) {
#if USE_TDM
            if (wid == 0) __builtin_amdgcn_s_wait_tensorcnt(0);  // DMA done before barrier
#else
            stage(knext, cur ^ 1);
#endif
        } else {
            stage(knext, cur ^ 1);
        }
        __syncthreads();
    }

    // ---- epilogue: biases + gate math + write ----
    const int col = bn0 + waveN * 16 + r;
    const float vb_ir = b_ir[col], vb_iz = b_iz[col], vb_in = b_in[col];
    const float vb_hr = b_hr[col], vb_hz = b_hz[col], vb_hn = b_hn[col];

#pragma unroll
    for (int s = 0; s < 2; ++s) {
#pragma unroll
        for (int v = 0; v < 8; ++v) {
            const int row = bm0 + waveM * 32 + s * 16 + v + 8 * half;
            const float ir  = acc[0][s][v] + vb_ir;
            const float iz  = acc[1][s][v] + vb_iz;
            const float inn = acc[2][s][v] + vb_in;
            const float hr  = acc[3][s][v] + vb_hr;
            const float hz  = acc[4][s][v] + vb_hz;
            const float hn  = acc[5][s][v] + vb_hn;
            const float rg = sigmoidf_fast(ir + hr);
            const float zg = sigmoidf_fast(iz + hz);
            const float ng = sigmoidf_fast(inn + rg * hn);   // source uses sigmoid, not tanh
            const float h  = hs[(size_t)row * HN + col];
            out[(size_t)row * HN + col] = (1.0f - zg) * h + zg * ng;
        }
    }
}

extern "C" void kernel_launch(void* const* d_in, const int* in_sizes, int n_in,
                              void* d_out, int out_size, void* d_ws, size_t ws_size,
                              hipStream_t stream) {
    (void)in_sizes; (void)n_in; (void)out_size;
    const float* inp  = (const float*)d_in[0];
    const float* hs   = (const float*)d_in[1];
    const float* w[6] = { (const float*)d_in[2], (const float*)d_in[4], (const float*)d_in[6],
                          (const float*)d_in[8], (const float*)d_in[10], (const float*)d_in[12] };
    const float* b_ir = (const float*)d_in[3];
    const float* b_iz = (const float*)d_in[5];
    const float* b_in = (const float*)d_in[7];
    const float* b_hr = (const float*)d_in[9];
    const float* b_hz = (const float*)d_in[11];
    const float* b_hn = (const float*)d_in[13];

    const dim3 grid(BATCH / BM, HN / BN);   // x = M tiles fastest -> weight slice hot in L2
    const dim3 block(256);

    const size_t inB = (size_t)BATCH * DK * 2;     // bf16 input bytes
    const size_t wB  = (size_t)HN * DK * 2;        // bf16 weight bytes (each)
    const size_t need = inB + 6 * wB;              // 256 MiB

    if (d_ws != nullptr && ws_size >= need) {
        uint8_t* ws = (uint8_t*)d_ws;
        uint32_t* inp_bf = (uint32_t*)ws;
        uint32_t* w_bf[6];
        for (int g = 0; g < 6; ++g) w_bf[g] = (uint32_t*)(ws + inB + (size_t)g * wB);

        // one-time fp32 -> bf16 conversions
        cvt_f32_bf16_kernel<<<4096, 256, 0, stream>>>(
            (const float4*)inp, (uint2*)inp_bf, (int)((size_t)BATCH * DK / 4));
        for (int g = 0; g < 6; ++g)
            cvt_f32_bf16_kernel<<<4096, 256, 0, stream>>>(
                (const float4*)w[g], (uint2*)w_bf[g], (int)((size_t)HN * DK / 4));

        gru_wmma_gemm<true><<<grid, block, 0, stream>>>(
            inp_bf, w_bf[0], w_bf[1], w_bf[2], w_bf[3], w_bf[4], w_bf[5],
            hs, b_ir, b_iz, b_in, b_hr, b_hz, b_hn, (float*)d_out);
    } else {
        gru_wmma_gemm<false><<<grid, block, 0, stream>>>(
            inp, w[0], w[1], w[2], w[3], w[4], w[5],
            hs, b_ir, b_iz, b_in, b_hr, b_hz, b_hn, (float*)d_out);
    }
}
